// NTXentLoss_5703716569566
// MI455X (gfx1250) — matrix operand order repaired and checked by reading, not compiled, not run on hardware
//
#include <hip/hip_runtime.h>
#include <stdint.h>

// NT-Xent fused: normalize -> bf16 WMMA similarity tiles (upper triangle) with
// in-register logsumexp epilogue -> final mean reduction.
// B=4096, D=1024, N=2B=8192, T=0.1 (logits bounded by 1/T=10 -> fixed LSE shift).
//
// Round 2: register-blocked GEMM. Each wave computes a 32x64 output block as
// 2x4 sub-tiles of 16x16 -> 8 WMMAs per K=32 step fed by 12 global_load_b128
// (1.5 loads/WMMA instead of 4.0). Block = 8 waves sharing the same 32 A-rows.

#define B_ROWS   4096
#define D_DIM    1024
#define N_ROWS   8192
#define INV_T    10.0f
#define SHIFT    10.0f
#define EPS_N    1e-8f

typedef __attribute__((ext_vector_type(16))) __bf16 bf16x16;
typedef __attribute__((ext_vector_type(8)))  float  f32x8;

__device__ __forceinline__ unsigned short f32_to_bf16_rne(float f) {
    union { float f; uint32_t u; } c; c.f = f;
    uint32_t u = c.u;
    uint32_t r = u + 0x7FFFu + ((u >> 16) & 1u);   // round-to-nearest-even
    return (unsigned short)(r >> 16);
}

// ---------------------------------------------------------------------------
// Kernel 1: per-row L2 normalize, convert to bf16, zero accumulators.
// One block (256 threads) per row; each thread handles 4 contiguous floats.
// ---------------------------------------------------------------------------
__global__ __launch_bounds__(256)
void ntxent_normalize(const float* __restrict__ zi,
                      const float* __restrict__ zj,
                      unsigned short* __restrict__ Rbf,
                      float* __restrict__ denom,
                      float* __restrict__ pos) {
    const int row = blockIdx.x;
    const int tid = threadIdx.x;
    const float* src = (row < B_ROWS) ? (zi + (size_t)row * D_DIM)
                                      : (zj + (size_t)(row - B_ROWS) * D_DIM);
    float4 x = ((const float4*)src)[tid];
    float ss = x.x * x.x + x.y * x.y + x.z * x.z + x.w * x.w;
    #pragma unroll
    for (int off = 16; off > 0; off >>= 1) ss += __shfl_xor(ss, off, 32);

    __shared__ float sred[8];
    __shared__ float sinv;
    if ((tid & 31) == 0) sred[tid >> 5] = ss;
    __syncthreads();
    if (tid == 0) {
        float t = 0.0f;
        #pragma unroll
        for (int i = 0; i < 8; ++i) t += sred[i];
        sinv = 1.0f / fmaxf(sqrtf(t), EPS_N);
        denom[row] = 0.0f;          // zero accumulators for this round
        pos[row]   = 0.0f;
    }
    __syncthreads();
    const float inv = sinv;

    ushort4 h;
    h.x = f32_to_bf16_rne(x.x * inv);
    h.y = f32_to_bf16_rne(x.y * inv);
    h.z = f32_to_bf16_rne(x.z * inv);
    h.w = f32_to_bf16_rne(x.w * inv);
    *(ushort4*)(Rbf + (size_t)row * D_DIM + 4 * tid) = h;
}

// ---------------------------------------------------------------------------
// Kernel 2: register-blocked similarity tiles via v_wmma_f32_16x16x32_bf16.
// Wave tile: 32 (M) x 64 (N) = 2x4 sub-tiles of 16x16.
// Block tile: 32 (M) x 512 (N) -> 8 waves share the same 32 A-rows.
// Grid: x = N/512 = 16, y = M/32 = 256. Upper triangle only (S symmetric):
// row-sums scatter to denom[m], column-sums to denom[n].
// Fused epilogue: exp(S-10) with diagonal mask, positive-pair capture on the
// nt16 == mt16 + 256 sub-tile diagonals.
// ---------------------------------------------------------------------------
__global__ __launch_bounds__(256)
void ntxent_sim_tiles(const unsigned short* __restrict__ R,
                      float* __restrict__ denom,
                      float* __restrict__ pos) {
    const int by = blockIdx.y;                   // M block: rows [32*by, 32*by+32)
    const int bx = blockIdx.x;                   // N block: cols [512*bx, 512*bx+512)
    const int mt16base = by << 1;                // first 16-row tile index (M)
    if ((bx << 5) + 31 < mt16base) return;       // whole block below diagonal

    const int tid  = threadIdx.x;
    const int lane = tid & 31;
    const int w    = tid >> 5;                   // wave id 0..7
    const int nt16base = (bx << 5) + (w << 2);   // wave's first 16-col tile (N)
    if (nt16base + 3 < mt16base) return;         // whole wave below diagonal

    const int hi  = lane >> 4;                   // half-wave selector
    const int l15 = lane & 15;

    // A layout (16-bit 16x32): lanes 0-15 hold M=l15, K in [0,8)+[16,24);
    //                          lanes 16-31 hold M=l15, K in [8,16)+[24,32).
    // B layout (32x16):        lanes 0-15 hold N=l15, K in [0,8)+[16,24);
    //                          lanes 16-31 hold N=l15, K in [8,16)+[24,32).
    const unsigned short* pa[2];
    const unsigned short* pb[4];
    #pragma unroll
    for (int i = 0; i < 2; ++i)
        pa[i] = R + (size_t)(((mt16base + i) << 4) + l15) * D_DIM + (hi << 3);
    #pragma unroll
    for (int j = 0; j < 4; ++j)
        pb[j] = R + (size_t)(((nt16base + j) << 4) + l15) * D_DIM + (hi << 3);

    f32x8 c[2][4] = {};
    union V16 { bf16x16 v; uint4 u[2]; };

    for (int k0 = 0; k0 < D_DIM; k0 += 32) {
        V16 a[2], b[4];
        #pragma unroll
        for (int i = 0; i < 2; ++i) {
            a[i].u[0] = *(const uint4*)(pa[i] + k0);
            a[i].u[1] = *(const uint4*)(pa[i] + k0 + 16);
        }
        #pragma unroll
        for (int j = 0; j < 4; ++j) {
            b[j].u[0] = *(const uint4*)(pb[j] + k0);
            b[j].u[1] = *(const uint4*)(pb[j] + k0 + 16);
        }
        #pragma unroll
        for (int i = 0; i < 2; ++i)
            #pragma unroll
            for (int j = 0; j < 4; ++j)
                c[i][j] = __builtin_amdgcn_wmma_f32_16x16x32_bf16(
                              false, a[i].v, false, b[j].v,
                              (short)0, c[i][j], false, false);
    }

    // ---- Epilogue ----
    // C/D layout: lane l15 holds N=l15; VGPR v holds M = v + 8*hi.
    float rowAcc[2][8];
    #pragma unroll
    for (int i = 0; i < 2; ++i)
        #pragma unroll
        for (int v = 0; v < 8; ++v) rowAcc[i][v] = 0.0f;

    #pragma unroll
    for (int j = 0; j < 4; ++j) {
        const int nt16 = nt16base + j;
        float cs = 0.0f;                         // column sums for this sub-col
        bool  anyCol = false;
        #pragma unroll
        for (int i = 0; i < 2; ++i) {
            const int mt16 = mt16base + i;
            if (nt16 < mt16) continue;           // sub-tile below diagonal
            const bool diagT = (nt16 == mt16);
            const bool posT  = (nt16 == mt16 + (B_ROWS >> 4));
            #pragma unroll
            for (int v = 0; v < 8; ++v) {
                const float s    = c[i][j][v] * INV_T;       // S[m][n]
                const int   mloc = v + (hi << 3);
                const bool  onD  = (mloc == l15);
                if (posT && onD) {                           // positive logit
                    const int mrow = (mt16 << 4) + mloc;
                    pos[mrow]          = s;
                    pos[mrow + B_ROWS] = s;                  // S symmetric
                }
                float ev = __expf(s - SHIFT);
                if (diagT && onD) ev = 0.0f;                 // mask self-sim
                rowAcc[i][v] += ev;
                if (!diagT) { cs += ev; anyCol = true; }     // mirror tile
            }
        }
        // Column sums: pair-sum across half-waves (M halves), then scatter.
        cs += __shfl_xor(cs, 16, 32);
        if (anyCol && lane < 16)
            __hip_atomic_fetch_add(&denom[(nt16 << 4) + lane], cs,
                                   __ATOMIC_RELAXED, __HIP_MEMORY_SCOPE_AGENT);
    }

    // Row sums: reduce over the 16 lanes of each half-wave (N dimension),
    // already accumulated across the 4 N-sub-tiles.
    #pragma unroll
    for (int i = 0; i < 2; ++i) {
        const int mt16 = mt16base + i;
        #pragma unroll
        for (int v = 0; v < 8; ++v) {
            float rs = rowAcc[i][v];
            rs += __shfl_xor(rs, 1, 32);
            rs += __shfl_xor(rs, 2, 32);
            rs += __shfl_xor(rs, 4, 32);
            rs += __shfl_xor(rs, 8, 32);
            if (l15 == v)                        // 2 lanes: rows v, v+8
                __hip_atomic_fetch_add(&denom[(mt16 << 4) + v + (hi << 3)], rs,
                                       __ATOMIC_RELAXED,
                                       __HIP_MEMORY_SCOPE_AGENT);
        }
    }
}

// ---------------------------------------------------------------------------
// Kernel 3: loss = mean( 10 + log(denom[i]) - pos[i] )
// ---------------------------------------------------------------------------
__global__ __launch_bounds__(256)
void ntxent_finalize(const float* __restrict__ denom,
                     const float* __restrict__ pos,
                     float* __restrict__ out) {
    const int tid = threadIdx.x;
    float acc = 0.0f;
    for (int i = tid; i < N_ROWS; i += 256)
        acc += logf(denom[i]) + SHIFT - pos[i];
    #pragma unroll
    for (int off = 16; off > 0; off >>= 1) acc += __shfl_xor(acc, off, 32);
    __shared__ float sred[8];
    if ((tid & 31) == 0) sred[tid >> 5] = acc;
    __syncthreads();
    if (tid == 0) {
        float t = 0.0f;
        #pragma unroll
        for (int i = 0; i < 8; ++i) t += sred[i];
        out[0] = t / (float)N_ROWS;
    }
}

// ---------------------------------------------------------------------------
extern "C" void kernel_launch(void* const* d_in, const int* in_sizes, int n_in,
                              void* d_out, int out_size, void* d_ws, size_t ws_size,
                              hipStream_t stream) {
    (void)in_sizes; (void)n_in; (void)out_size; (void)ws_size;
    const float* zi = (const float*)d_in[0];
    const float* zj = (const float*)d_in[1];
    float* out = (float*)d_out;

    // Workspace layout: [Rbf16: 8192*1024*2 = 16 MB][denom: 32 KB][pos: 32 KB]
    unsigned short* Rbf   = (unsigned short*)d_ws;
    float*          denom = (float*)((char*)d_ws + (size_t)N_ROWS * D_DIM * 2);
    float*          pos   = denom + N_ROWS;

    ntxent_normalize<<<dim3(N_ROWS), dim3(256), 0, stream>>>(zi, zj, Rbf, denom, pos);
    // N/512 = 16 column blocks, M/32 = 256 row blocks
    ntxent_sim_tiles<<<dim3(16, 256), dim3(256), 0, stream>>>(Rbf, denom, pos);
    ntxent_finalize<<<dim3(1), dim3(256), 0, stream>>>(denom, pos, out);
}
